// OptimizedBertSelfAttention_76622216561335
// MI455X (gfx1250) — compile-verified, matrix operand order checked by reading, not run on hardware
//
#include <hip/hip_runtime.h>

typedef __attribute__((ext_vector_type(16))) _Float16 v16h;
typedef __attribute__((ext_vector_type(8)))  _Float16 v8h;
typedef __attribute__((ext_vector_type(4)))  _Float16 v4h;
typedef __attribute__((ext_vector_type(8)))  float    v8f;

#define NB    8
#define SEQ   1024
#define HID   1024
#define NHEAD 16
#define DHEAD 64

__device__ __forceinline__ v8f wmma_f16(v16h a, v16h b, v8f c) {
  // D = A(16x32 f16) * B(32x16 f16) + C(16x16 f32)
  return __builtin_amdgcn_wmma_f32_16x16x32_f16(false, a, false, b, (short)0, c,
                                                false, false);
}

// Load an A-layout (or B-as-A-of-Bt) fragment from a row-major f16 matrix.
// Per CDNA5 ISA 16-bit A layout: lanes 0-15 hold row M=lane, K = {kb..kb+7, kb+16..kb+23};
// lanes 16-31 hold row M=lane-16, K = {kb+8..kb+15, kb+24..kb+31}.
__device__ __forceinline__ v16h load_frag(const _Float16* base, int row, int ld,
                                          int kbase, int lane) {
  const int half = (lane & 16) ? 8 : 0;
  const _Float16* p = base + (size_t)row * ld + kbase + half;
  v8h c0 = *(const v8h*)(p);
  v8h c1 = *(const v8h*)(p + 16);
  v16h a;
#pragma unroll
  for (int j = 0; j < 8; ++j) { a[j] = c0[j]; a[j + 8] = c1[j]; }
  return a;
}

// ---------------------------------------------------------------------------
// fp32 -> fp16 conversion (bandwidth kernel)
// ---------------------------------------------------------------------------
__global__ __launch_bounds__(256) void f32_to_f16_kernel(
    const float* __restrict__ src, _Float16* __restrict__ dst, int n) {
  int i = (blockIdx.x * blockDim.x + threadIdx.x) * 4;
  if (i + 3 < n) {
    float4 v = *(const float4*)(src + i);
    v4h o;
    o[0] = (_Float16)v.x; o[1] = (_Float16)v.y;
    o[2] = (_Float16)v.z; o[3] = (_Float16)v.w;
    *(v4h*)(dst + i) = o;
  } else {
    for (int j = i; j < n; ++j) dst[j] = (_Float16)src[j];
  }
}

// ---------------------------------------------------------------------------
// Fused QKV projection: Y = Xh @ W^T + bias.
// grid = (M/64, N/64, 3).  128 threads = 4 waves; wave owns 16 rows x 64 cols.
// z==0 -> Q [B,h,S,d], z==1 -> K [B,h,S,d], z==2 -> V transposed [B,h,d,S].
// ---------------------------------------------------------------------------
__global__ __launch_bounds__(128) void qkv_gemm_kernel(
    const _Float16* __restrict__ Xh, const _Float16* __restrict__ Wh3,
    const float* __restrict__ bq, const float* __restrict__ bk,
    const float* __restrict__ bv,
    _Float16* __restrict__ Qb, _Float16* __restrict__ Kb,
    _Float16* __restrict__ Vt) {
  const int mt = blockIdx.x;          // 0..127, 64 rows each (rows = B*S)
  const int nt = blockIdx.y;          // 0..15,  64 cols each (cols = HID)
  const int z  = blockIdx.z;          // 0:Q 1:K 2:V
  const int wave = threadIdx.x >> 5;
  const int lane = threadIdx.x & 31;
  const int lane15 = lane & 15;
  const int hi = (lane & 16) ? 8 : 0;

  const _Float16* W = Wh3 + (size_t)z * HID * HID;
  const float* bias = (z == 0) ? bq : ((z == 1) ? bk : bv);

  const int row = mt * 64 + wave * 16 + lane15;  // A-fragment row for this lane

  v8f acc[4];
#pragma unroll
  for (int t = 0; t < 4; ++t)
#pragma unroll
    for (int r = 0; r < 8; ++r) acc[t][r] = 0.0f;

  for (int k = 0; k < HID; k += 32) {
    v16h a = load_frag(Xh, row, HID, k, lane);
#pragma unroll
    for (int t = 0; t < 4; ++t) {
      v16h bf = load_frag(W, nt * 64 + t * 16 + lane15, HID, k, lane);
      acc[t] = wmma_f16(a, bf, acc[t]);
    }
  }

  // Epilogue: C layout -> lane holds fixed column n, rows m = wavebase+hi+r.
#pragma unroll
  for (int t = 0; t < 4; ++t) {
    const int n = nt * 64 + t * 16 + lane15;
    const float bn = bias[n];
    const int head = n >> 6;
    const int dh = n & 63;
#pragma unroll
    for (int r = 0; r < 8; ++r) {
      const int m = mt * 64 + wave * 16 + hi + r;
      const int b = m >> 10;
      const int s = m & 1023;
      const float y = acc[t][r] + bn;
      if (z == 2) {
        Vt[(((size_t)(b * NHEAD + head) * DHEAD) + dh) * SEQ + s] = (_Float16)y;
      } else {
        _Float16* dst = (z == 0) ? Qb : Kb;
        dst[(((size_t)(b * NHEAD + head) * SEQ) + s) * DHEAD + dh] = (_Float16)y;
      }
    }
  }
}

// ---------------------------------------------------------------------------
// Flash attention: one block per (q-tile of 64, head, batch).
// 128 threads = 4 waves; wave owns 16 q-rows. Online softmax over 16 k-tiles.
// ---------------------------------------------------------------------------
__global__ __launch_bounds__(128) void flash_attn_kernel(
    const _Float16* __restrict__ Qb, const _Float16* __restrict__ Kb,
    const _Float16* __restrict__ Vt, const float* __restrict__ mask,
    float* __restrict__ out) {
  const int qt = blockIdx.x;   // 0..15
  const int h  = blockIdx.y;   // 0..15
  const int b  = blockIdx.z;   // 0..7
  const int wave = threadIdx.x >> 5;
  const int lane = threadIdx.x & 31;
  const int lane15 = lane & 15;
  const int hi = (lane & 16) ? 8 : 0;

  constexpr float SCALE = 0.125f;            // 1/sqrt(64)
  constexpr float LOG2E = 1.44269504088896f;

  const _Float16* Q = Qb + (size_t)(b * NHEAD + h) * SEQ * DHEAD;
  const _Float16* K = Kb + (size_t)(b * NHEAD + h) * SEQ * DHEAD;
  const _Float16* V = Vt + (size_t)(b * NHEAD + h) * DHEAD * SEQ;  // [d][s]
  const float* mrow = mask + (size_t)b * SEQ;

  // per-wave staging buffer for P (C layout -> A layout), padded to kill bank conflicts
  __shared__ __align__(16) _Float16 ldsP[4][16][72];
  const _Float16* myP = &ldsP[wave][0][0];

  const int qrow = qt * 64 + wave * 16 + lane15;
  const v16h aq0 = load_frag(Q, qrow, DHEAD, 0, lane);
  const v16h aq1 = load_frag(Q, qrow, DHEAD, 32, lane);

  v8f o[4];
  float m_run[8], l_run[8];
#pragma unroll
  for (int t = 0; t < 4; ++t)
#pragma unroll
    for (int r = 0; r < 8; ++r) o[t][r] = 0.0f;
#pragma unroll
  for (int r = 0; r < 8; ++r) { m_run[r] = -__builtin_inff(); l_run[r] = 0.0f; }

  for (int kt = 0; kt < 16; ++kt) {
    const int kb = kt * 64;

    if (kt < 15) {  // gfx1250 global_prefetch on next K/V tiles
      __builtin_prefetch(K + (size_t)(kb + 64 + lane15 * 4) * DHEAD, 0, 1);
      __builtin_prefetch(V + (size_t)(lane15 * 4) * SEQ + kb + 64, 0, 1);
    }

    // ---- S = Q K^T ----
    v8f sf[4];
#pragma unroll
    for (int t = 0; t < 4; ++t)
#pragma unroll
      for (int r = 0; r < 8; ++r) sf[t][r] = 0.0f;

#pragma unroll
    for (int t = 0; t < 4; ++t) {
      v16h bk0 = load_frag(K, kb + t * 16 + lane15, DHEAD, 0, lane);
      v16h bk1 = load_frag(K, kb + t * 16 + lane15, DHEAD, 32, lane);
      sf[t] = wmma_f16(aq0, bk0, sf[t]);
      sf[t] = wmma_f16(aq1, bk1, sf[t]);
    }

    // scale + additive mask (mask fixed per lane's column)
#pragma unroll
    for (int t = 0; t < 4; ++t) {
      const float mv = mrow[kb + t * 16 + lane15];
#pragma unroll
      for (int r = 0; r < 8; ++r) sf[t][r] = sf[t][r] * SCALE + mv;
    }

    // ---- online softmax ----
    float newm[8], alpha[8];
#pragma unroll
    for (int r = 0; r < 8; ++r) {
      float v = fmaxf(fmaxf(sf[0][r], sf[1][r]), fmaxf(sf[2][r], sf[3][r]));
#pragma unroll
      for (int off = 1; off < 16; off <<= 1) v = fmaxf(v, __shfl_xor(v, off, 32));
      newm[r] = fmaxf(m_run[r], v);
      alpha[r] = exp2f((m_run[r] - newm[r]) * LOG2E);
      m_run[r] = newm[r];
    }
#pragma unroll
    for (int t = 0; t < 4; ++t)
#pragma unroll
      for (int r = 0; r < 8; ++r)
        sf[t][r] = exp2f((sf[t][r] - newm[r]) * LOG2E);
#pragma unroll
    for (int r = 0; r < 8; ++r) {
      float v = sf[0][r] + sf[1][r] + sf[2][r] + sf[3][r];
#pragma unroll
      for (int off = 1; off < 16; off <<= 1) v += __shfl_xor(v, off, 32);
      l_run[r] = l_run[r] * alpha[r] + v;
    }

    // rescale running output
#pragma unroll
    for (int t = 0; t < 4; ++t)
#pragma unroll
      for (int r = 0; r < 8; ++r) o[t][r] *= alpha[r];

    // ---- P (C layout) -> LDS (row-major) -> A fragments ----
#pragma unroll
    for (int t = 0; t < 4; ++t)
#pragma unroll
      for (int r = 0; r < 8; ++r)
        ldsP[wave][hi + r][t * 16 + lane15] = (_Float16)sf[t][r];
    // same-wave DS ops are in-order: no barrier needed for private buffer
    v16h ap0 = load_frag(myP, lane15, 72, 0, lane);
    v16h ap1 = load_frag(myP, lane15, 72, 32, lane);

    // ---- O += P V  (B fragments = rows of Vt, contiguous b128 loads) ----
#pragma unroll
    for (int t = 0; t < 4; ++t) {
      v16h bv0 = load_frag(V, t * 16 + lane15, SEQ, kb + 0, lane);
      v16h bv1 = load_frag(V, t * 16 + lane15, SEQ, kb + 32, lane);
      o[t] = wmma_f16(ap0, bv0, o[t]);
      o[t] = wmma_f16(ap1, bv1, o[t]);
    }
  }

  // ---- epilogue: out[b, s, h*64 + d] = O / l ----
#pragma unroll
  for (int r = 0; r < 8; ++r) {
    const float inv = 1.0f / l_run[r];
    const int srow = qt * 64 + wave * 16 + hi + r;
    float* orow = out + ((size_t)(b * SEQ + srow)) * HID + h * DHEAD;
#pragma unroll
    for (int t = 0; t < 4; ++t) orow[t * 16 + lane15] = o[t][r] * inv;
  }
}

// ---------------------------------------------------------------------------
// Workspace layout (bytes):
//   Xh  f16 [8192][1024]        @ 0         (16,777,216)
//   Wh3 f16 [3][1024][1024]     @ 16777216  ( 6,291,456)
//   Qb  f16 [8][16][1024][64]   @ 23068672  (16,777,216)
//   Kb  f16 [8][16][1024][64]   @ 39845888  (16,777,216)
//   Vt  f16 [8][16][64][1024]   @ 56623104  (16,777,216)  -> total ~70 MB
// ---------------------------------------------------------------------------
extern "C" void kernel_launch(void* const* d_in, const int* in_sizes, int n_in,
                              void* d_out, int out_size, void* d_ws, size_t ws_size,
                              hipStream_t stream) {
  (void)in_sizes; (void)n_in; (void)out_size; (void)ws_size;
  const float* x    = (const float*)d_in[0];
  const float* mask = (const float*)d_in[1];
  const float* Wq   = (const float*)d_in[2];
  const float* bq   = (const float*)d_in[3];
  const float* Wk   = (const float*)d_in[4];
  const float* bk   = (const float*)d_in[5];
  const float* Wv   = (const float*)d_in[6];
  const float* bv   = (const float*)d_in[7];
  float* out = (float*)d_out;

  char* ws = (char*)d_ws;
  _Float16* Xh  = (_Float16*)(ws + 0);
  _Float16* Wh3 = (_Float16*)(ws + 16777216);
  _Float16* Qb  = (_Float16*)(ws + 23068672);
  _Float16* Kb  = (_Float16*)(ws + 39845888);
  _Float16* Vt  = (_Float16*)(ws + 56623104);

  const int nX = NB * SEQ * HID;   // 8,388,608
  const int nW = HID * HID;        // 1,048,576

  f32_to_f16_kernel<<<(nX / 4 + 255) / 256, 256, 0, stream>>>(x, Xh, nX);
  f32_to_f16_kernel<<<(nW / 4 + 255) / 256, 256, 0, stream>>>(Wq, Wh3 + (size_t)0 * nW, nW);
  f32_to_f16_kernel<<<(nW / 4 + 255) / 256, 256, 0, stream>>>(Wk, Wh3 + (size_t)1 * nW, nW);
  f32_to_f16_kernel<<<(nW / 4 + 255) / 256, 256, 0, stream>>>(Wv, Wh3 + (size_t)2 * nW, nW);

  qkv_gemm_kernel<<<dim3(128, 16, 3), 128, 0, stream>>>(Xh, Wh3, bq, bk, bv,
                                                        Qb, Kb, Vt);

  flash_attn_kernel<<<dim3(16, 16, 8), 128, 0, stream>>>(Qb, Kb, Vt, mask, out);
}
